// HGCN_UI_19146964205953
// MI455X (gfx1250) — matrix operand order repaired and checked by reading, not compiled, not run on hardware
//
#include <hip/hip_runtime.h>
#include <hip/hip_bf16.h>

typedef __attribute__((ext_vector_type(16))) __bf16 v16bf;
typedef __attribute__((ext_vector_type(8)))  float  v8f;

#define DIM  128
#define KDIM 256

// ---------------- elementwise helpers (float4 vectorized) ----------------
__global__ void __launch_bounds__(256) zero4_kernel(float4* __restrict__ p, long long n4) {
  long long i = (long long)blockIdx.x * blockDim.x + threadIdx.x;
  if (i < n4) p[i] = make_float4(0.f, 0.f, 0.f, 0.f);
}

__global__ void __launch_bounds__(256) copy4_kernel(float4* __restrict__ dst,
                                                    const float4* __restrict__ src,
                                                    long long n4) {
  long long i = (long long)blockIdx.x * blockDim.x + threadIdx.x;
  if (i < n4) dst[i] = src[i];
}

__global__ void __launch_bounds__(256) add4_kernel(float4* __restrict__ dst,
                                                   const float4* __restrict__ src,
                                                   long long n4) {
  long long i = (long long)blockIdx.x * blockDim.x + threadIdx.x;
  if (i < n4) {
    float4 d = dst[i];
    float4 s = src[i];
    d.x += s.x; d.y += s.y; d.z += s.z; d.w += s.w;
    dst[i] = d;
  }
}

// ---------------- SpMM scatter: out[sidx[e]] += vals[e] * src[gidx[e]] ----------------
// One wave per nnz edge: 32 lanes x float4 = 128 dims. Gather sources are
// L2-resident (25-51MB << 192MB L2); scatter uses native f32 atomics which
// resolve at L2 (global_atomic_add_f32, no CAS loop).
__global__ void __launch_bounds__(256)
spmm_kernel(float* __restrict__ out, const float* __restrict__ src,
            const float* __restrict__ vals,
            const int* __restrict__ gidx, const int* __restrict__ sidx,
            int nnz) {
  int edge = blockIdx.x * 8 + (threadIdx.x >> 5);
  if (edge >= nnz) return;
  int lane = threadIdx.x & 31;
  int g = gidx[edge];
  int s = sidx[edge];
  float v = vals[edge];
  const float4* sp = (const float4*)(src + (size_t)g * DIM);
  float* op = out + (size_t)s * DIM + lane * 4;
  float4 x = sp[lane];
  __hip_atomic_fetch_add(op + 0, v * x.x, __ATOMIC_RELAXED, __HIP_MEMORY_SCOPE_AGENT);
  __hip_atomic_fetch_add(op + 1, v * x.y, __ATOMIC_RELAXED, __HIP_MEMORY_SCOPE_AGENT);
  __hip_atomic_fetch_add(op + 2, v * x.z, __ATOMIC_RELAXED, __HIP_MEMORY_SCOPE_AGENT);
  __hip_atomic_fetch_add(op + 3, v * x.w, __ATOMIC_RELAXED, __HIP_MEMORY_SCOPE_AGENT);
}

// ---------------- W pre-pack: f32 [256,128] -> bf16 in WMMA B-fragment order --------
// Fragment element mapping (16x16x32 bf16 B, 32x16 KxN):
//   lane L (n = nt*16 + L%16, half h = L/16), VGPR j covers K = kt*32 + h*16 + 2j (+1).
// Packed layout: Bpk[((kt*8 + nt)*32 + lane)*16 + 2j+{0,1}] so each lane's whole
// fragment is one contiguous 32-byte (v16bf) load, coalesced across lanes.
__global__ void __launch_bounds__(256)
pack_w_kernel(const float* __restrict__ W, __bf16* __restrict__ Bpk) {
  int t = blockIdx.x * blockDim.x + threadIdx.x;  // (kt, nt, lane) flattened
  if (t >= (KDIM / 32) * (DIM / 16) * 32) return;
  int lane = t & 31;
  int nt   = (t >> 5) & 7;
  int kt   = t >> 8;
  int lm = lane & 15, h = lane >> 4;
  __bf16* o = Bpk + (size_t)t * 16;
#pragma unroll
  for (int j = 0; j < 8; ++j) {
    int kb = kt * 32 + (h << 4) + (j << 1);
    o[2 * j]     = (__bf16)W[(size_t)kb * DIM + nt * 16 + lm];
    o[2 * j + 1] = (__bf16)W[(size_t)(kb + 1) * DIM + nt * 16 + lm];
  }
}

// ---------------- GEMM: msg = [nm | nm*edge] @ W + b ; fedge += msg ----------------
// Block = 256 threads = 8 waves. Block owns a 16-row M tile; wave w owns the
// 16-col N tile [16w, 16w+16). A tile (16x256, concat built on the fly) staged
// in LDS as bf16 once per block; B fragments are single wide loads from the
// pre-packed bf16 weights; 8 x v_wmma_f32_16x16x32_bf16 per wave.
__global__ void __launch_bounds__(256)
gemm_msg_kernel(const float* __restrict__ nm,     // [M,128] node_msg
                const float* __restrict__ edge,   // [M,128] edge (gate)
                const v16bf* __restrict__ Bpk,    // packed bf16 W fragments
                const float* __restrict__ bias,   // [128]
                float* __restrict__ msg,          // [M,128] out
                float* __restrict__ fedge,        // [M,128] accumulator
                int M) {
  __shared__ __bf16 sA[16][KDIM + 2];  // pad to 258 bf16 -> conflict-free column reads
  const int m0  = blockIdx.x * 16;
  const int tid = threadIdx.x;

  // Cooperative A-tile staging: thread loads 16 consecutive K of one row.
  {
    int mr  = tid >> 4;          // 0..15 row in tile
    int kc0 = (tid & 15) * 16;   // 0..240 start col
    int row = m0 + mr;
    if (row >= M) row = M - 1;   // clamp (keeps EXEC full)
    const float* nmr = nm   + (size_t)row * DIM;
    const float* edr = edge + (size_t)row * DIM;
#pragma unroll
    for (int kk = 0; kk < 16; ++kk) {
      int k = kc0 + kk;
      float v = (k < DIM) ? nmr[k] : nmr[k - DIM] * edr[k - DIM];
      sA[mr][k] = (__bf16)v;
    }
  }
  __syncthreads();

  const int lane = tid & 31;
  const int wave = tid >> 5;   // 0..7 -> N tile
  const int n0   = wave * 16;
  const int lm   = lane & 15;
  const int h    = lane >> 4;  // lane half

  v8f acc = {};
#pragma unroll
  for (int kt = 0; kt < KDIM / 32; ++kt) {
    v16bf a;
#pragma unroll
    for (int j = 0; j < 8; ++j) {
      // A 16x32 bf16 fragment: lanes 0-15 = M rows; VGPR j covers K pair
      // (half h -> K base 16h, j*2 within).
      int ka = kt * 32 + (h << 4) + (j << 1);
      a[2 * j]     = sA[lm][ka];
      a[2 * j + 1] = sA[lm][ka + 1];
    }
    // B fragment: one contiguous 32B load per lane from packed weights.
    v16bf b = Bpk[(kt * 8 + wave) * 32 + lane];
    acc = __builtin_amdgcn_wmma_f32_16x16x32_bf16(
        /*neg_a=*/false, a, /*neg_b=*/false, b,
        /*c_mod=*/(short)0, acc, /*reuse_a=*/false, /*reuse_b=*/false);
  }

  // Epilogue: D VGPR r -> M = m0 + r + 8h, N = n0 + lm. Fuse bias + fedge accum.
  float bv = bias[n0 + lm];
#pragma unroll
  for (int r = 0; r < 8; ++r) {
    int row = m0 + r + (h << 3);
    if (row < M) {
      size_t idx = (size_t)row * DIM + n0 + lm;
      float v = acc[r] + bv;
      msg[idx] = v;
      fedge[idx] += v;
    }
  }
}

// ---------------- host orchestration ----------------
static inline int div_up_ll(long long a, int b) { return (int)((a + b - 1) / b); }

extern "C" void kernel_launch(void* const* d_in, const int* in_sizes, int n_in,
                              void* d_out, int out_size, void* d_ws, size_t ws_size,
                              hipStream_t stream) {
  (void)n_in; (void)out_size; (void)ws_size;
  const float* user_emb = (const float*)d_in[0];
  const float* item_emb = (const float*)d_in[1];
  const float* vals     = (const float*)d_in[2];
  const float* agg_w    = (const float*)d_in[3];
  const float* agg_b    = (const float*)d_in[4];
  const int*   rows     = (const int*)d_in[5];
  const int*   cols     = (const int*)d_in[6];

  const int n_users = in_sizes[0] / DIM;   // 100000
  const int n_items = in_sizes[1] / DIM;   // 50000
  const int nnz     = in_sizes[2];         // 2000000
  const int layers  = in_sizes[4] / DIM;   // 3

  const long long nU = (long long)n_users * DIM;  // user-side elems
  const long long nI = (long long)n_items * DIM;  // item-side elems

  float* out        = (float*)d_out;
  float* final_node = out;                  // [n_items, DIM]
  float* final_edge = out + nI;             // [n_users, DIM]

  float* ws    = (float*)d_ws;
  float* nm    = ws;                        // [n_users, DIM]
  float* msg0  = nm + nU;                   // [n_users, DIM]
  float* msg1  = msg0 + nU;                 // [n_users, DIM]
  float* nodeB = msg1 + nU;                 // [n_items, DIM]
  __bf16* wpk  = (__bf16*)(nodeB + nI);     // [layers][256*128] packed bf16 W

  const int TB = 256;
  const int wfrag = (KDIM / 32) * (DIM / 16) * 32;  // fragments-lanes per layer

  // Pre-pack each layer's weights into WMMA B-fragment order (bf16).
  for (int i = 0; i < layers; ++i) {
    pack_w_kernel<<<div_up_ll(wfrag, TB), TB, 0, stream>>>(
        agg_w + (size_t)i * 2 * DIM * DIM, wpk + (size_t)i * KDIM * DIM);
  }

  // final_node = item_emb ; final_edge = user_emb
  copy4_kernel<<<div_up_ll(nI / 4, TB), TB, 0, stream>>>(
      (float4*)final_node, (const float4*)item_emb, nI / 4);
  copy4_kernel<<<div_up_ll(nU / 4, TB), TB, 0, stream>>>(
      (float4*)final_edge, (const float4*)user_emb, nU / 4);

  const float* node_src = item_emb;  // layer 0 node
  for (int i = 0; i < layers; ++i) {
    float* msg_cur        = (i & 1) ? msg1 : msg0;
    const float* edge_src = (i == 0) ? user_emb : ((i & 1) ? msg0 : msg1);
    const v16bf* Wi = (const v16bf*)(wpk + (size_t)i * KDIM * DIM);
    const float* bi = agg_b + (size_t)i * DIM;

    // nm = H @ node  (scatter by rows, gather by cols)
    zero4_kernel<<<div_up_ll(nU / 4, TB), TB, 0, stream>>>((float4*)nm, nU / 4);
    spmm_kernel<<<div_up_ll(nnz, 8), TB, 0, stream>>>(nm, node_src, vals, cols, rows, nnz);

    // msg = [nm | nm*edge] @ W + b ; final_edge += msg   (WMMA)
    gemm_msg_kernel<<<div_up_ll(n_users, 16), TB, 0, stream>>>(
        nm, edge_src, Wi, bi, msg_cur, final_edge, n_users);

    // node = H^T @ msg  (scatter by cols, gather by rows)
    zero4_kernel<<<div_up_ll(nI / 4, TB), TB, 0, stream>>>((float4*)nodeB, nI / 4);
    spmm_kernel<<<div_up_ll(nnz, 8), TB, 0, stream>>>(nodeB, msg_cur, vals, rows, cols, nnz);

    // final_node += node
    add4_kernel<<<div_up_ll(nI / 4, TB), TB, 0, stream>>>(
        (float4*)final_node, (const float4*)nodeB, nI / 4);

    node_src = nodeB;
  }
}
